// GraphGAT_60284160966756
// MI455X (gfx1250) — matrix operand order repaired and checked by reading, not compiled, not run on hardware
//
#include <hip/hip_runtime.h>
#include <hip/hip_bf16.h>
#include <hip/hip_fp16.h>

typedef __attribute__((ext_vector_type(16))) _Float16 v16h;
typedef __attribute__((ext_vector_type(8)))  float    v8f;
typedef __attribute__((ext_vector_type(8)))  unsigned v8u;

#define NNODES 50000
#define NEDGES 800000
#define INDIM  128
#define HDIM   64      // HEADS*HID for layer1, OUT_DIM for layer2
#define NTOT   192     // 3 * 64 (src|dst|val projections concatenated)
#define NTILEN 12      // 192 / 16
#define MTILES 3125    // 50000 / 16

// ---------------- helpers ----------------

__device__ __forceinline__ unsigned f2ord(float f) {
    unsigned u = __float_as_uint(f);
    return (u & 0x80000000u) ? ~u : (u | 0x80000000u);
}
__device__ __forceinline__ float ord2f(unsigned u) {
    return (u & 0x80000000u) ? __uint_as_float(u & 0x7fffffffu) : __uint_as_float(~u);
}
__device__ __forceinline__ float wave_sum(float v) {
#pragma unroll
    for (int off = 16; off > 0; off >>= 1) v += __shfl_xor(v, off, 32);
    return v;
}

// ---------------- init kernels ----------------

__global__ void zero_f32(float* p, int n) {
    int i = blockIdx.x * blockDim.x + threadIdx.x;
    if (i < n) p[i] = 0.0f;
}
__global__ void init_neg_inf_ord(unsigned* p, int n) {
    int i = blockIdx.x * blockDim.x + threadIdx.x;
    if (i < n) p[i] = 0x007fffffu;   // f2ord(-inf)
}

// ---------------- LayerNorm(x) -> f16, 128 dims, 1 wave / node ----------------

__global__ void ln128_kernel(const float* __restrict__ x,
                             const float* __restrict__ g, const float* __restrict__ b,
                             _Float16* __restrict__ xn, int n) {
    int node = blockIdx.x * blockDim.y + threadIdx.y;
    if (node >= n) return;
    int lane = threadIdx.x;
    const float* row = x + (size_t)node * INDIM;
    float v[4]; float s = 0.f;
#pragma unroll
    for (int j = 0; j < 4; ++j) { v[j] = row[lane + 32 * j]; s += v[j]; }
    s = wave_sum(s);
    float m = s * (1.0f / 128.0f);
    float q = 0.f;
#pragma unroll
    for (int j = 0; j < 4; ++j) { float d = v[j] - m; q += d * d; }
    q = wave_sum(q);
    float r = rsqrtf(q * (1.0f / 128.0f) + 1e-3f);
    _Float16* orow = xn + (size_t)node * INDIM;
#pragma unroll
    for (int j = 0; j < 4; ++j) {
        int c = lane + 32 * j;
        orow[c] = (_Float16)((v[j] - m) * r * g[c] + b[c]);
    }
}

// -------- ELU + LayerNorm -> f16, 64 dims, 1 wave / node --------

__global__ void eluln64_kernel(const float* __restrict__ hin,
                               const float* __restrict__ g, const float* __restrict__ b,
                               _Float16* __restrict__ hout, int n) {
    int node = blockIdx.x * blockDim.y + threadIdx.y;
    if (node >= n) return;
    int lane = threadIdx.x;
    const float* row = hin + (size_t)node * HDIM;
    float v0 = row[lane], v1 = row[lane + 32];
    v0 = v0 > 0.f ? v0 : (expf(v0) - 1.0f);
    v1 = v1 > 0.f ? v1 : (expf(v1) - 1.0f);
    float s = wave_sum(v0 + v1);
    float m = s * (1.0f / 64.0f);
    float d0 = v0 - m, d1 = v1 - m;
    float q = wave_sum(d0 * d0 + d1 * d1);
    float r = rsqrtf(q * (1.0f / 64.0f) + 1e-3f);
    _Float16* orow = hout + (size_t)node * HDIM;
    orow[lane]      = (_Float16)(d0 * r * g[lane]      + b[lane]);
    orow[lane + 32] = (_Float16)(d1 * r * g[lane + 32] + b[lane + 32]);
}

// -------- pack weights [K,64]x3 -> WMMA B-fragment order (f16 pairs) --------
// Bp[((tn*nK + kt)*8 + v)*32 + lane] packs B[K][N] with
//   N = tn*16 + (lane&15),  K = kt*32 + 2*v + (lane>>4)*8 + (v>>2)*8  (+1 in hi half)

__global__ void pack_w3_kernel(const float* __restrict__ Ws, const float* __restrict__ Wd,
                               const float* __restrict__ Wv, unsigned* __restrict__ Bp, int K) {
    int idx = blockIdx.x * blockDim.x + threadIdx.x;
    int nK = K >> 5;
    int total = NTILEN * nK * 256;
    if (idx >= total) return;
    int lane = idx & 31;
    int v    = (idx >> 5) & 7;
    int kt   = (idx >> 8) % nK;
    int tn   = (idx >> 8) / nK;
    int col  = tn * 16 + (lane & 15);
    int k0   = kt * 32 + 2 * v + ((lane >> 4) << 3) + ((v >> 2) << 3);
    const float* W = (col < 64) ? Ws : ((col < 128) ? Wd : Wv);
    int cc = col & 63;
    _Float16 lo = (_Float16)W[(size_t)k0 * 64 + cc];
    _Float16 hi = (_Float16)W[(size_t)(k0 + 1) * 64 + cc];
    unsigned ul = __builtin_bit_cast(unsigned short, lo);
    unsigned uh = __builtin_bit_cast(unsigned short, hi);
    Bp[idx] = ul | (uh << 16);
}

// -------- WMMA GEMM: A f16 [M,K] row-major x Bp packed -> D f32 [M,192] --------
// 1 wave per 16x16 tile, K-loop of 32 per v_wmma_f32_16x16x32_f16.

__global__ __launch_bounds__(32) void gemm_wmma_kernel(const _Float16* __restrict__ A,
                                                       const unsigned* __restrict__ Bp,
                                                       float* __restrict__ D, int K) {
    int nK = K >> 5;
    int tn = blockIdx.x % NTILEN;
    int tm = blockIdx.x / NTILEN;
    int lane = threadIdx.x;
    int hf = lane >> 4;                 // lane half (0: lanes 0-15, 1: 16-31)
    int row = tm * 16 + (lane & 15);
    const unsigned* arow = (const unsigned*)(A + (size_t)row * K);   // f16 pairs
    const unsigned* bbase = Bp + (((size_t)tn * nK) << 8);
    v8f acc = {};
    for (int kt = 0; kt < nK; ++kt) {
        v8u au, bu;
        int kb = kt << 4;               // pair-index base of this K-step
#pragma unroll
        for (int v = 0; v < 4; ++v) {
            au[v]     = arow[kb + v + 4 * hf];        // K = kt*32 + 2v   + 8*hf
            au[v + 4] = arow[kb + 8 + v + 4 * hf];    // K = kt*32+16+2v + 8*hf
        }
        const unsigned* bp = bbase + (kt << 8) + lane;
#pragma unroll
        for (int v = 0; v < 8; ++v) bu[v] = bp[v << 5];
        v16h a = __builtin_bit_cast(v16h, au);
        v16h b = __builtin_bit_cast(v16h, bu);
        acc = __builtin_amdgcn_wmma_f32_16x16x32_f16(false, a, false, b,
                                                     (short)0, acc, false, false);
    }
    float* drow = D + (size_t)(tm * 16 + 8 * hf) * NTOT + tn * 16 + (lane & 15);
#pragma unroll
    for (int r = 0; r < 8; ++r) drow[(size_t)r * NTOT] = acc[r];
}

// -------- GATv2 edge scores, 2 heads of 32 dims, 1 wave / edge --------

__global__ void score_h2_kernel(const float* __restrict__ Hm, const int* __restrict__ src,
                                const int* __restrict__ dst, const float* __restrict__ avec,
                                float* __restrict__ scoreOut, unsigned* __restrict__ segmax,
                                int E) {
    int e = blockIdx.x * blockDim.y + threadIdx.y;
    if (e >= E) return;
    int lane = threadIdx.x;
    const float* hs = Hm + (size_t)src[e] * NTOT;        // cols   0..63  (W_src proj)
    const float* hd = Hm + (size_t)dst[e] * NTOT + 64;   // cols  64..127 (W_dst proj)
    float m0 = hs[lane] + hd[lane];
    m0 = m0 > 0.f ? m0 : 0.2f * m0;
    float s0 = m0 * avec[lane];                          // head 0, dim lane
    float m1 = hs[lane + 32] + hd[lane + 32];
    m1 = m1 > 0.f ? m1 : 0.2f * m1;
    float s1 = m1 * avec[lane + 32];                     // head 1, dim lane
    s0 = wave_sum(s0);
    s1 = wave_sum(s1);
    if (lane == 0) {
        int dn = dst[e];
        scoreOut[(size_t)e * 2]     = s0;
        scoreOut[(size_t)e * 2 + 1] = s1;
        atomicMax(segmax + (size_t)dn * 2,     f2ord(s0));
        atomicMax(segmax + (size_t)dn * 2 + 1, f2ord(s1));
    }
}

// -------- GATv2 edge scores, 1 head of 64 dims, 1 wave / edge --------

__global__ void score_h1_kernel(const float* __restrict__ Hm, const int* __restrict__ src,
                                const int* __restrict__ dst, const float* __restrict__ avec,
                                float* __restrict__ scoreOut, unsigned* __restrict__ segmax,
                                int E) {
    int e = blockIdx.x * blockDim.y + threadIdx.y;
    if (e >= E) return;
    int lane = threadIdx.x;
    const float* hs = Hm + (size_t)src[e] * NTOT;
    const float* hd = Hm + (size_t)dst[e] * NTOT + 64;
    float m0 = hs[lane] + hd[lane];
    m0 = m0 > 0.f ? m0 : 0.2f * m0;
    float s = m0 * avec[lane];
    float m1 = hs[lane + 32] + hd[lane + 32];
    m1 = m1 > 0.f ? m1 : 0.2f * m1;
    s += m1 * avec[lane + 32];
    s = wave_sum(s);
    if (lane == 0) {
        scoreOut[e] = s;
        atomicMax(segmax + dst[e], f2ord(s));
    }
}

// -------- exp(score - segmax[dst]) in place + segment sum --------

__global__ void expsum_kernel(float* __restrict__ sc, const int* __restrict__ dst,
                              const unsigned* __restrict__ segmax, float* __restrict__ segsum,
                              int E, int heads) {
    int idx = blockIdx.x * blockDim.x + threadIdx.x;
    if (idx >= E * heads) return;
    int e = idx / heads, h = idx - e * heads;
    int seg = dst[e] * heads + h;
    float es = expf(sc[idx] - ord2f(segmax[seg]));
    sc[idx] = es;
    atomicAdd(segsum + seg, es);
}

// -------- attn = exp_s / (segsum[dst] + 1e-9) in place --------

__global__ void norm_kernel(float* __restrict__ sc, const int* __restrict__ dst,
                            const float* __restrict__ segsum, int E, int heads) {
    int idx = blockIdx.x * blockDim.x + threadIdx.x;
    if (idx >= E * heads) return;
    int e = idx / heads, h = idx - e * heads;
    sc[idx] = sc[idx] / (segsum[dst[e] * heads + h] + 1e-9f);
}

// -------- aggregate: accum[dst, c] += attn[e,h(c)] * Hval[src, c] --------

__global__ void agg_kernel(const float* __restrict__ attn, const float* __restrict__ Hm,
                           const int* __restrict__ src, const int* __restrict__ dst,
                           float* __restrict__ accum, int E, int heads) {
    int idx = blockIdx.x * blockDim.x + threadIdx.x;
    if (idx >= E * 64) return;
    int e = idx >> 6, c = idx & 63;
    int h = (c >> 5) & (heads - 1);
    float a = attn[(size_t)e * heads + h];
    float val = a * Hm[(size_t)src[e] * NTOT + 128 + c];   // cols 128..191 = W_val proj
    atomicAdd(accum + (size_t)dst[e] * 64 + c, val);
}

// ---------------- launch ----------------

extern "C" void kernel_launch(void* const* d_in, const int* in_sizes, int n_in,
                              void* d_out, int out_size, void* d_ws, size_t ws_size,
                              hipStream_t stream) {
    const float* x    = (const float*)d_in[0];
    const int*   ei   = (const int*)d_in[1];
    const float* Ws1  = (const float*)d_in[2];
    const float* Wd1  = (const float*)d_in[3];
    const float* Wv1  = (const float*)d_in[4];
    const float* a1   = (const float*)d_in[5];
    const float* g_in = (const float*)d_in[6];
    const float* b_in = (const float*)d_in[7];
    const float* g_ln = (const float*)d_in[8];
    const float* b_ln = (const float*)d_in[9];
    const float* Ws2  = (const float*)d_in[10];
    const float* Wd2  = (const float*)d_in[11];
    const float* Wv2  = (const float*)d_in[12];
    const float* a2   = (const float*)d_in[13];
    const int* src = ei;
    const int* dst = ei + NEDGES;

    // workspace layout
    char* ws = (char*)d_ws;
    _Float16* xn   = (_Float16*)ws;                         // 50000*128*2 = 12.8MB (reused as h1n f16)
    float*    Hm   = (float*)(ws + 12800000);               // 50000*192*4 = 38.4MB (H1 then H2)
    unsigned* W1p  = (unsigned*)(ws + 12800000 + 38400000); // 12288 * 4B
    unsigned* W2p  = W1p + 12288;                           // 6144 * 4B
    unsigned* segmax = (unsigned*)((char*)(W2p + 6144));    // 50000*2*4 = 400KB
    float*    segsum = (float*)((char*)segmax + 400000);    // 400KB
    float*    h1pre  = (float*)((char*)segsum + 400000);    // 50000*64*4 = 12.8MB
    float*    score2 = (float*)((char*)h1pre + 12800000);   // 800000*4 = 3.2MB

    float* outp  = (float*)d_out;            // [50000, 64]
    float* attn1 = outp + NNODES * 64;       // [800000, 2]

    dim3 w8(32, 8);

    // ---- layer 1 ----
    ln128_kernel<<<(NNODES + 7) / 8, w8, 0, stream>>>(x, g_in, b_in, xn, NNODES);
    pack_w3_kernel<<<(NTILEN * 4 * 256 + 255) / 256, 256, 0, stream>>>(Ws1, Wd1, Wv1, W1p, 128);
    pack_w3_kernel<<<(NTILEN * 2 * 256 + 255) / 256, 256, 0, stream>>>(Ws2, Wd2, Wv2, W2p, 64);
    gemm_wmma_kernel<<<MTILES * NTILEN, 32, 0, stream>>>(xn, W1p, Hm, 128);

    init_neg_inf_ord<<<(NNODES * 2 + 255) / 256, 256, 0, stream>>>(segmax, NNODES * 2);
    zero_f32<<<(NNODES * 2 + 255) / 256, 256, 0, stream>>>(segsum, NNODES * 2);
    zero_f32<<<(NNODES * 64 + 255) / 256, 256, 0, stream>>>(h1pre, NNODES * 64);
    zero_f32<<<(NNODES * 64 + 255) / 256, 256, 0, stream>>>(outp, NNODES * 64);

    score_h2_kernel<<<(NEDGES + 7) / 8, w8, 0, stream>>>(Hm, src, dst, a1, attn1, segmax, NEDGES);
    expsum_kernel<<<(NEDGES * 2 + 255) / 256, 256, 0, stream>>>(attn1, dst, segmax, segsum, NEDGES, 2);
    norm_kernel<<<(NEDGES * 2 + 255) / 256, 256, 0, stream>>>(attn1, dst, segsum, NEDGES, 2);
    agg_kernel<<<(NEDGES * 64 + 255) / 256, 256, 0, stream>>>(attn1, Hm, src, dst, h1pre, NEDGES, 2);

    // ---- layer 2 ----
    eluln64_kernel<<<(NNODES + 7) / 8, w8, 0, stream>>>(h1pre, g_ln, b_ln, xn, NNODES);
    gemm_wmma_kernel<<<MTILES * NTILEN, 32, 0, stream>>>(xn, W2p, Hm, 64);

    init_neg_inf_ord<<<(NNODES + 255) / 256, 256, 0, stream>>>(segmax, NNODES);
    zero_f32<<<(NNODES + 255) / 256, 256, 0, stream>>>(segsum, NNODES);

    score_h1_kernel<<<(NEDGES + 7) / 8, w8, 0, stream>>>(Hm, src, dst, a2, score2, segmax, NEDGES);
    expsum_kernel<<<(NEDGES + 255) / 256, 256, 0, stream>>>(score2, dst, segmax, segsum, NEDGES, 1);
    norm_kernel<<<(NEDGES + 255) / 256, 256, 0, stream>>>(score2, dst, segsum, NEDGES, 1);
    agg_kernel<<<(NEDGES * 64 + 255) / 256, 256, 0, stream>>>(score2, Hm, src, dst, outp, NEDGES, 1);
}
